// FusedAttention_37984690766201
// MI455X (gfx1250) — matrix operand order repaired and checked by reading, not compiled, not run on hardware
//
#include <hip/hip_runtime.h>
#include <hip/hip_bf16.h>

typedef __attribute__((ext_vector_type(16))) __bf16 v16bf;
typedef __attribute__((ext_vector_type(8)))  __bf16 v8bf;
typedef __attribute__((ext_vector_type(4)))  __bf16 v4bf;
typedef __attribute__((ext_vector_type(8)))  float  v8f;

// ---------------------------------------------------------------------------
// CDNA5 async LDS staging: GLOBAL_LOAD_ASYNC_TO_LDS_B128 (ASYNCcnt-tracked).
// LDS destination address = low 32 bits of the generic pointer to __shared__.
// ---------------------------------------------------------------------------
__device__ __forceinline__ void async_load_b128(unsigned lds_off, const void* gaddr) {
    asm volatile("global_load_async_to_lds_b128 %0, %1, off"
                 :: "v"(lds_off), "v"((unsigned long long)gaddr) : "memory");
}
__device__ __forceinline__ void wait_async0() {
    asm volatile("s_wait_asynccnt 0" ::: "memory");
}
__device__ __forceinline__ unsigned lds_off(const void* p) {
    return (unsigned)(unsigned long long)p;
}

// ---------------------------------------------------------------------------
// Fragment helpers for wave32 WMMA 16x16x32 bf16 layouts.
// A (16x32): lanes 0-15 hold M=0..15; K chunks {half*8+0..7},{16+half*8+0..7}.
// B (32x16): lane n = lane&15 is column; same K chunks as A.
// Both reduce to two 16-byte contiguous loads per lane.
// ---------------------------------------------------------------------------
__device__ __forceinline__ v16bf load_frag(const __bf16* p0, const __bf16* p1) {
    union { v16bf v; v8bf h[2]; } u;
    u.h[0] = *(const v8bf*)p0;
    u.h[1] = *(const v8bf*)p1;
    return u.v;
}
__device__ __forceinline__ v8f wmma_bf16(v16bf a, v16bf b, v8f c) {
    return __builtin_amdgcn_wmma_f32_16x16x32_bf16(false, a, false, b, (short)0, c,
                                                   false, false);
}

// ---------------------------------------------------------------------------
// fp32 -> bf16 bulk convert (pre-pass so all GEMM staging is pure async copy)
// ---------------------------------------------------------------------------
__global__ __launch_bounds__(256) void to_bf16_k(const float* __restrict__ in,
                                                 __bf16* __restrict__ out, long n4) {
    long t = (long)blockIdx.x * 256 + threadIdx.x;
    if (t >= n4) return;
    float4 f = ((const float4*)in)[t];
    v4bf o = {(__bf16)f.x, (__bf16)f.y, (__bf16)f.z, (__bf16)f.w};
    ((v4bf*)out)[t] = o;
}

// ---------------------------------------------------------------------------
// GEMM: C[m][n] = sum_k A[m][k] * W[n][k], A/W bf16 K-major, C = float or bf16.
// 256 threads = 8 waves (4x2), block tile 128x128, BK=32, double-buffered LDS
// filled by async B128 copies; 8 WMMAs per wave per k-step.
// ---------------------------------------------------------------------------
#define G_BM 128
#define G_BN 128
#define G_BK 32
#define G_LD 40   // bf16 row stride (80B: 16B aligned, bank-skewed)

template <typename OutT>
__global__ __launch_bounds__(256) void gemm_wmma(const __bf16* __restrict__ A,
                                                 const __bf16* __restrict__ W,
                                                 OutT* __restrict__ C,
                                                 int M, int N, int K) {
    __shared__ __attribute__((aligned(16))) __bf16 As[2][G_BM * G_LD];
    __shared__ __attribute__((aligned(16))) __bf16 Bs[2][G_BN * G_LD];

    const int tid  = threadIdx.x;
    const int lane = tid & 31, wave = tid >> 5;
    const int half = lane >> 4, l16 = lane & 15;
    const int wm = wave >> 1, wn = wave & 1;        // 4x2 waves: tile 32(M) x 64(N)
    const int bm = blockIdx.y * G_BM, bn = blockIdx.x * G_BN;

    // staging map: 128 rows x 4 x 16B chunks each matrix; 2 chunks per thread
    const int sr = tid >> 1;
    const int sc = (tid & 1) * 2;

    auto stage = [&](int buf, int k0) {
        #pragma unroll
        for (int i = 0; i < 2; ++i) {
            const int c8 = (sc + i) * 8;
            async_load_b128(lds_off(&As[buf][sr * G_LD + c8]),
                            &A[(size_t)(bm + sr) * K + k0 + c8]);
            async_load_b128(lds_off(&Bs[buf][sr * G_LD + c8]),
                            &W[(size_t)(bn + sr) * K + k0 + c8]);
        }
    };

    v8f acc[2][4] = {};
    stage(0, 0);
    wait_async0();
    __syncthreads();

    const int nt = K / G_BK;
    for (int t = 0; t < nt; ++t) {
        const int cur = t & 1;
        if (t + 1 < nt) stage(cur ^ 1, (t + 1) * G_BK);

        v16bf af[2], bfr[4];
        #pragma unroll
        for (int i = 0; i < 2; ++i) {
            const __bf16* ar = &As[cur][(wm * 32 + i * 16 + l16) * G_LD];
            af[i] = load_frag(ar + half * 8, ar + 16 + half * 8);
        }
        #pragma unroll
        for (int j = 0; j < 4; ++j) {
            const __bf16* br = &Bs[cur][(wn * 64 + j * 16 + l16) * G_LD];
            bfr[j] = load_frag(br + half * 8, br + 16 + half * 8);
        }
        #pragma unroll
        for (int i = 0; i < 2; ++i)
            #pragma unroll
            for (int j = 0; j < 4; ++j)
                acc[i][j] = wmma_bf16(af[i], bfr[j], acc[i][j]);

        wait_async0();        // next tile landed (issued before compute)
        __syncthreads();      // safe to overwrite buffer cur next iteration
    }

    #pragma unroll
    for (int i = 0; i < 2; ++i)
        #pragma unroll
        for (int j = 0; j < 4; ++j)
            #pragma unroll
            for (int r = 0; r < 8; ++r) {
                const int row = bm + wm * 32 + i * 16 + half * 8 + r;
                const int col = bn + wn * 64 + j * 16 + l16;
                C[(size_t)row * N + col] = (OutT)acc[i][j][r];
            }
}

// ---------------------------------------------------------------------------
// RoPE + head scatter.  qkv bf16 [B,S,3072] ->
//   qb bf16 [B,16,S,128] rotated, kb bf16 [B,4,S,128] rotated,
//   vb bf16 [B,4,128,S]  TRANSPOSED plain convert (d-major for async staging).
// ---------------------------------------------------------------------------
__global__ __launch_bounds__(256) void rope_scatter(const __bf16* __restrict__ qkv,
                                                    __bf16* __restrict__ qb,
                                                    __bf16* __restrict__ kb,
                                                    __bf16* __restrict__ vb,
                                                    int S) {
    long t = (long)blockIdx.x * blockDim.x + threadIdx.x;
    int  i    = (int)(t & 63);
    long r1   = t >> 6;
    int  unit = (int)(r1 % 24);
    long r2   = r1 / 24;
    int  s    = (int)(r2 % S);
    int  b    = (int)(r2 / S);
    const __bf16* row = qkv + ((size_t)b * S + s) * 3072;

    if (unit < 20) {
        float inv = __expf(-(float)(2 * i) * (9.210340371976184f / 128.0f));
        float ang = (float)s * inv;
        float sn, cs;
        __sincosf(ang, &sn, &cs);
        if (unit < 16) {
            float a = (float)row[unit * 128 + 2 * i];
            float c = (float)row[unit * 128 + 2 * i + 1];
            __bf16* o = qb + (((size_t)b * 16 + unit) * S + s) * 128 + 2 * i;
            o[0] = (__bf16)(a * cs - c * sn);
            o[1] = (__bf16)(a * sn + c * cs);
        } else {
            int kh = unit - 16;
            float a = (float)row[2048 + kh * 128 + 2 * i];
            float c = (float)row[2048 + kh * 128 + 2 * i + 1];
            __bf16* o = kb + (((size_t)b * 4 + kh) * S + s) * 128 + 2 * i;
            o[0] = (__bf16)(a * cs - c * sn);
            o[1] = (__bf16)(a * sn + c * cs);
        }
    } else {
        int vh = unit - 20;
        // transposed: vb[b][vh][d][s]
        __bf16* o = vb + (((size_t)b * 4 + vh) * 128 + 2 * i) * (size_t)S + s;
        o[0] = row[2560 + vh * 128 + 2 * i];
        o[S] = row[2560 + vh * 128 + 2 * i + 1];
    }
}

// ---------------------------------------------------------------------------
// Causal GQA flash attention with WMMA + async double-buffered KV staging.
// Grid (S/128, 16, B); 256 threads = 8 waves; each wave owns 16 q rows.
// ---------------------------------------------------------------------------
#define A_HD   128
#define A_QT   128
#define A_KT   32
#define A_KLD  136
#define A_VLD  40
#define A_PLD  40

__global__ __launch_bounds__(256) void attn_wmma(const __bf16* __restrict__ Qg,
                                                 const __bf16* __restrict__ Kg,
                                                 const __bf16* __restrict__ Vtg,
                                                 __bf16* __restrict__ Og,
                                                 int S) {
    __shared__ __attribute__((aligned(16))) __bf16 Ks[2][A_KT * A_KLD];
    __shared__ __attribute__((aligned(16))) __bf16 Vs[2][A_HD * A_VLD];  // [d][kv]
    __shared__ __attribute__((aligned(16))) __bf16 Ps[8 * 16 * A_PLD];

    const int tid  = threadIdx.x;
    const int lane = tid & 31, wave = tid >> 5;
    const int half = lane >> 4, l16 = lane & 15;
    const int q0 = blockIdx.x * A_QT;
    const int h  = blockIdx.y, b = blockIdx.z;
    const int hkv = h >> 2;

    const __bf16* Qbase = Qg  + (((size_t)b * 16 + h)  * S) * A_HD;
    const __bf16* Kbase = Kg  + (((size_t)b * 4 + hkv) * S) * A_HD;
    const __bf16* Vtb   = Vtg + (((size_t)b * 4 + hkv) * A_HD) * S;   // [128][S]

    auto stageKV = [&](int buf, int kv0) {
        #pragma unroll
        for (int it = 0; it < 2; ++it) {
            const int c = tid + it * 256;             // 512 chunks each matrix
            const int kr = c >> 4, kc8 = (c & 15) * 8;
            async_load_b128(lds_off(&Ks[buf][kr * A_KLD + kc8]),
                            &Kbase[(size_t)(kv0 + kr) * A_HD + kc8]);
            const int vr = c >> 2, vc8 = (c & 3) * 8;
            async_load_b128(lds_off(&Vs[buf][vr * A_VLD + vc8]),
                            &Vtb[(size_t)vr * S + kv0 + vc8]);
        }
    };

    // Preload this wave's 16 Q rows (4 A-fragments covering hd=128)
    const __bf16* qrow = Qbase + (size_t)(q0 + wave * 16 + l16) * A_HD;
    v16bf qf[4];
    #pragma unroll
    for (int c = 0; c < 4; ++c)
        qf[c] = load_frag(qrow + c * 32 + half * 8, qrow + c * 32 + 16 + half * 8);

    v8f   ao[8] = {};
    float mrow[8], lrow[8];
    #pragma unroll
    for (int r = 0; r < 8; ++r) { mrow[r] = -1e30f; lrow[r] = 0.0f; }

    const float scale = 0.08838834764831845f;         // 1/sqrt(128)
    const int   wrow0 = q0 + wave * 16;
    const int   ntile = (q0 + A_QT) / A_KT;           // causal bound

    stageKV(0, 0);
    wait_async0();
    __syncthreads();

    for (int t = 0; t < ntile; ++t) {
        const int cur = t & 1;
        const int kv0 = t * A_KT;
        if (t + 1 < ntile) stageKV(cur ^ 1, kv0 + A_KT);

        if (kv0 <= wrow0 + 15) {                      // wave-uniform live-tile test
            // ---- scores: two 16x16 tiles over 32 kv columns ----
            v8f sc[2] = {};
            #pragma unroll
            for (int sub = 0; sub < 2; ++sub) {
                const __bf16* kr = &Ks[cur][(sub * 16 + l16) * A_KLD];
                #pragma unroll
                for (int c = 0; c < 4; ++c) {
                    v16bf kfr = load_frag(kr + c * 32 + half * 8,
                                          kr + c * 32 + 16 + half * 8);
                    sc[sub] = wmma_bf16(qf[c], kfr, sc[sub]);
                }
            }

            const bool needmask = (kv0 + A_KT - 1) > wrow0;
            float p0v[8], p1v[8];
            #pragma unroll
            for (int r = 0; r < 8; ++r) {
                const int qr = wrow0 + half * 8 + r;
                float s0 = sc[0][r] * scale;
                float s1 = sc[1][r] * scale;
                if (needmask) {
                    if (kv0 + l16 > qr)      s0 = -1e9f;
                    if (kv0 + 16 + l16 > qr) s1 = -1e9f;
                }
                float tmax = fmaxf(s0, s1);
                tmax = fmaxf(tmax, __shfl_xor(tmax, 1));
                tmax = fmaxf(tmax, __shfl_xor(tmax, 2));
                tmax = fmaxf(tmax, __shfl_xor(tmax, 4));
                tmax = fmaxf(tmax, __shfl_xor(tmax, 8));
                const float mn = fmaxf(mrow[r], tmax);
                const float sf = __expf(mrow[r] - mn);
                const float p0 = __expf(s0 - mn);
                const float p1 = __expf(s1 - mn);
                float rs = p0 + p1;
                rs += __shfl_xor(rs, 1);
                rs += __shfl_xor(rs, 2);
                rs += __shfl_xor(rs, 4);
                rs += __shfl_xor(rs, 8);
                lrow[r] = lrow[r] * sf + rs;
                mrow[r] = mn;
                #pragma unroll
                for (int dt = 0; dt < 8; ++dt) ao[dt][r] *= sf;
                p0v[r] = p0; p1v[r] = p1;
            }

            // ---- P: C-layout -> A-layout through per-wave LDS scratch ----
            __bf16* pw = &Ps[wave * 16 * A_PLD];
            #pragma unroll
            for (int r = 0; r < 8; ++r) {
                pw[(half * 8 + r) * A_PLD + l16]      = (__bf16)p0v[r];
                pw[(half * 8 + r) * A_PLD + 16 + l16] = (__bf16)p1v[r];
            }
            asm volatile("s_wait_dscnt 0" ::: "memory");   // same-wave LDS RAW
            const __bf16* prow = pw + l16 * A_PLD;
            v16bf pf = load_frag(prow + half * 8, prow + 16 + half * 8);

            // ---- O += P x V over 8 d-chunks ----
            #pragma unroll
            for (int dt = 0; dt < 8; ++dt) {
                const __bf16* vr = &Vs[cur][(dt * 16 + l16) * A_VLD];
                v16bf vfr = load_frag(vr + half * 8, vr + 16 + half * 8);
                ao[dt] = wmma_bf16(pf, vfr, ao[dt]);
            }
        }

        wait_async0();     // executed by ALL waves every iteration
        __syncthreads();
    }

    #pragma unroll
    for (int dt = 0; dt < 8; ++dt)
        #pragma unroll
        for (int r = 0; r < 8; ++r) {
            const int srow = wrow0 + half * 8 + r;
            float o = ao[dt][r] / lrow[r];
            Og[((size_t)b * S + srow) * 2048 + h * 128 + dt * 16 + l16] = (__bf16)o;
        }
}

// ---------------------------------------------------------------------------
// Host orchestration.
// ---------------------------------------------------------------------------
extern "C" void kernel_launch(void* const* d_in, const int* in_sizes, int n_in,
                              void* d_out, int out_size, void* d_ws, size_t ws_size,
                              hipStream_t stream) {
    const float* x     = (const float*)d_in[0];   // [B,S,D] fp32
    const float* w_qkv = (const float*)d_in[1];   // [3072,2048] fp32
    const float* w_o   = (const float*)d_in[2];   // [2048,2048] fp32
    float*       out   = (float*)d_out;

    const int B = 2, S = 2048, D = 2048, HQ = 16, HKV = 4, HD = 128;
    const int M  = B * S;                          // 4096
    const int NQ = (HQ + 2 * HKV) * HD;            // 3072

    char*  ws  = (char*)d_ws;
    size_t off = 0;
    __bf16* xb   = (__bf16*)(ws + off); off += (size_t)M * D * 2;              // 16 MB
    __bf16* wqb  = (__bf16*)(ws + off); off += (size_t)NQ * D * 2;             // 12 MB
    __bf16* wob  = (__bf16*)(ws + off); off += (size_t)D * D * 2;              //  8 MB
    __bf16* qkvb = (__bf16*)(ws + off); off += (size_t)M * NQ * 2;             // 24 MB
    __bf16* qb   = (__bf16*)(ws + off); off += (size_t)B * HQ  * S * HD * 2;   // 16 MB
    __bf16* kb   = (__bf16*)(ws + off); off += (size_t)B * HKV * S * HD * 2;   //  4 MB
    __bf16* vb   = (__bf16*)(ws + off); off += (size_t)B * HKV * S * HD * 2;   //  4 MB
    __bf16* ob   = (__bf16*)(ws + off); off += (size_t)M * D * 2;              // 16 MB

    // 0) one-shot fp32 -> bf16 conversions (makes all tile staging pure DMA)
    long n4x = (long)M * D / 4, n4q = (long)NQ * D / 4, n4o = (long)D * D / 4;
    to_bf16_k<<<(int)((n4x + 255) / 256), 256, 0, stream>>>(x, xb, n4x);
    to_bf16_k<<<(int)((n4q + 255) / 256), 256, 0, stream>>>(w_qkv, wqb, n4q);
    to_bf16_k<<<(int)((n4o + 255) / 256), 256, 0, stream>>>(w_o, wob, n4o);

    // 1) fused QKV projection
    gemm_wmma<__bf16><<<dim3(NQ / G_BN, M / G_BM), 256, 0, stream>>>(xb, wqb, qkvb, M, NQ, D);

    // 2) RoPE + scatter (V written d-major for async-friendly attention tiles)
    rope_scatter<<<(B * S * 24 * 64) / 256, 256, 0, stream>>>(qkvb, qb, kb, vb, S);

    // 3) causal GQA flash attention
    attn_wmma<<<dim3(S / A_QT, HQ, B), 256, 0, stream>>>(qb, kb, vb, ob, S);

    // 4) output projection straight into d_out (fp32)
    gemm_wmma<float><<<dim3(D / G_BN, M / G_BM), 256, 0, stream>>>(ob, wob, out, M, D, D);
}
